// LocalManifoldFFNLayer_56573309224688
// MI455X (gfx1250) — matrix operand order repaired and checked by reading, not compiled, not run on hardware
//
#include <hip/hip_runtime.h>
#include <math.h>

// ---------------------------------------------------------------------------
// Types for CDNA5 WMMA (wave32): v_wmma_f32_16x16x32_bf16
// ---------------------------------------------------------------------------
typedef __bf16 bf16_t;
typedef __attribute__((ext_vector_type(16))) __bf16 v16bf;
typedef __attribute__((ext_vector_type(8)))  __bf16 v8bf;
typedef __attribute__((ext_vector_type(8)))  float  v8f;
typedef __attribute__((ext_vector_type(4)))  unsigned int v4u;
typedef __attribute__((ext_vector_type(8)))  int v8i;
typedef __attribute__((ext_vector_type(4)))  int v4i;

#if __has_builtin(__builtin_amdgcn_tensor_load_to_lds) && __has_builtin(__builtin_amdgcn_s_wait_tensorcnt)
#define USE_TDM 1
#else
#define USE_TDM 0
#endif

static __device__ __forceinline__ v8f zero_v8f() {
  v8f z;
  for (int i = 0; i < 8; ++i) z[i] = 0.0f;
  return z;
}
static __device__ __forceinline__ v8bf zero_v8bf() {
  v8bf z;
  for (int i = 0; i < 8; ++i) z[i] = (__bf16)0.0f;
  return z;
}
static __device__ __forceinline__ v16bf cat16(v8bf a, v8bf b) {
  return __builtin_shufflevector(a, b, 0,1,2,3,4,5,6,7,8,9,10,11,12,13,14,15);
}
static __device__ __forceinline__ v8bf ldv8(const bf16_t* p) {
  return *(const v8bf*)p;
}
static __device__ __forceinline__ float gelu_exact(float x) {
  return 0.5f * x * (1.0f + erff(x * 0.70710678118654752f));
}
static __device__ __forceinline__ v8f wmma_bf16(v16bf a, v16bf b, v8f c) {
  return __builtin_amdgcn_wmma_f32_16x16x32_bf16(false, a, false, b, (short)0, c, false, false);
}

#if USE_TDM
// Issue one TDM 2D-tile load: 64 rows x 32 bf16 elements, row stride = strideElems,
// from global byte address ga into LDS byte offset ldsOff (dense 64x32 destination).
// D# layout per CDNA5 ISA section 8.3-8.6. 6-arg builtin form (clang-23 lane):
// (uint32x4 g0, int32x8 g1, int32x4 g2, int32x4 g3, int32x8 g4, i32 cpol)
static __device__ __forceinline__ void tdm_load_tile_64x32_bf16(unsigned long long ga,
                                                                unsigned int ldsOff,
                                                                int strideElems) {
  unsigned int galo = (unsigned int)(ga & 0xFFFFFFFFull);
  unsigned int gahi = (unsigned int)((ga >> 32) & 0x01FFFFFFull) | (2u << 30); // type=2 (image)
  galo = (unsigned int)__builtin_amdgcn_readfirstlane((int)galo);
  gahi = (unsigned int)__builtin_amdgcn_readfirstlane((int)gahi);
  ldsOff = (unsigned int)__builtin_amdgcn_readfirstlane((int)ldsOff);
  int strd = __builtin_amdgcn_readfirstlane(strideElems);

  v4u g0;
  g0[0] = 1u;        // count=1, user descriptor
  g0[1] = ldsOff;    // lds_addr
  g0[2] = galo;      // global_addr[31:0]
  g0[3] = gahi;      // global_addr[56:32] | type<<30

  v8i g1;
  g1[0] = (1 << 16);                 // wg_mask=0, data_size=1 (2 bytes), no flags
  g1[1] = (32 << 16);                // tensor_dim0[15:0]=32 in bits[31:16]
  g1[2] = (64 << 16);                // tensor_dim0 hi=0 | tensor_dim1[15:0]=64
  g1[3] = (32 << 16);                // tensor_dim1 hi=0 | tile_dim0=32
  g1[4] = 64;                        // tile_dim1=64, tile_dim2=0
  g1[5] = strd;                      // tensor_dim0_stride[31:0] (elements)
  g1[6] = 0;                         // stride hi, dim1_stride lo
  g1[7] = 0;                         // dim1_stride hi
  v4i g2; g2[0] = 0; g2[1] = 0; g2[2] = 0; g2[3] = 0;
  v4i g3 = g2;
  v8i g4; for (int i = 0; i < 8; ++i) g4[i] = 0;
  __builtin_amdgcn_tensor_load_to_lds(g0, g1, g2, g3, g4, 0);
}
#endif

// Problem constants
#define D_MODEL 1024
#define D_FF    4096
#define D_NEURON 128
#define HEAD_DIM 16
#define K_TOP   32
#define LN_EPS  1e-5f

// ---------------------------------------------------------------------------
// fp32 -> bf16 cast (weights / tables)
// ---------------------------------------------------------------------------
__global__ __launch_bounds__(256) void cast_f32_bf16(const float* __restrict__ src,
                                                     bf16_t* __restrict__ dst, int n) {
  int i = blockIdx.x * 256 + threadIdx.x;
  if (i < n) dst[i] = (bf16_t)src[i];
}

// ---------------------------------------------------------------------------
// LayerNorm over D_MODEL, output bf16 (router input). One block per token.
// ---------------------------------------------------------------------------
__global__ __launch_bounds__(256) void ln_kernel(const float* __restrict__ x,
                                                 const float* __restrict__ g,
                                                 const float* __restrict__ b,
                                                 bf16_t* __restrict__ xn) {
  __shared__ float red[256];
  const int t = blockIdx.x, tid = threadIdx.x;
  const float* xr = x + (size_t)t * D_MODEL;
  float4 v = *(const float4*)(xr + tid * 4);
  float s = v.x + v.y + v.z + v.w;
  red[tid] = s; __syncthreads();
  for (int k = 128; k > 0; k >>= 1) { if (tid < k) red[tid] += red[tid + k]; __syncthreads(); }
  float mu = red[0] * (1.0f / D_MODEL);
  __syncthreads();
  float dx = v.x - mu, dy = v.y - mu, dz = v.z - mu, dw = v.w - mu;
  red[tid] = dx*dx + dy*dy + dz*dz + dw*dw; __syncthreads();
  for (int k = 128; k > 0; k >>= 1) { if (tid < k) red[tid] += red[tid + k]; __syncthreads(); }
  float rs = rsqrtf(red[0] * (1.0f / D_MODEL) + LN_EPS);
  float4 gg = *(const float4*)(g + tid * 4);
  float4 bb = *(const float4*)(b + tid * 4);
  bf16_t* o = xn + (size_t)t * D_MODEL + tid * 4;
  o[0] = (bf16_t)(dx * rs * gg.x + bb.x);
  o[1] = (bf16_t)(dy * rs * gg.y + bb.y);
  o[2] = (bf16_t)(dz * rs * gg.z + bb.z);
  o[3] = (bf16_t)(dw * rs * gg.w + bb.w);
}

// ---------------------------------------------------------------------------
// Tiled WMMA GEMM: C[M,N] = A[M,K] @ B[N,K]^T (row-row dot), bf16 in, f32 acc.
// mode 0: store f32. mode 1: store gelu(x) as bf16.
// Block: 128 threads (4 waves). Tile 64x64, K-step 32.
// Tile staging via TDM (tensor_load_to_lds, TENSORcnt) when available.
// ---------------------------------------------------------------------------
__global__ __launch_bounds__(128) void gemm_bf16_kernel(const bf16_t* __restrict__ A,
                                                        const bf16_t* __restrict__ B,
                                                        float* __restrict__ Cf,
                                                        bf16_t* __restrict__ Cb,
                                                        int M, int N, int K, int mode) {
  __shared__ __align__(16) bf16_t sA[64 * 32];
  __shared__ __align__(16) bf16_t sB[64 * 32];
  const int tid = threadIdx.x, wave = tid >> 5, lane = tid & 31;
  const int mlo = lane & 15, kh = lane >> 4;
  const int bm = blockIdx.y * 64, bn = blockIdx.x * 64;

  v8f acc0 = zero_v8f(), acc1 = zero_v8f(), acc2 = zero_v8f(), acc3 = zero_v8f();

#if USE_TDM
  const unsigned int ldsA = (unsigned int)(size_t)(void*)sA;
  const unsigned int ldsB = (unsigned int)(size_t)(void*)sB;
#else
  const int row0 = tid >> 2, off0 = (tid & 3) * 8;
  const int row1 = row0 + 32, off1 = off0;
#endif

  for (int k0 = 0; k0 < K; k0 += 32) {
#if USE_TDM
    if (wave == 0) {
      tdm_load_tile_64x32_bf16((unsigned long long)(size_t)&A[(size_t)bm * K + k0], ldsA, K);
      tdm_load_tile_64x32_bf16((unsigned long long)(size_t)&B[(size_t)bn * K + k0], ldsB, K);
      __builtin_amdgcn_s_wait_tensorcnt(0);
    }
    __syncthreads();
#else
    *(v8bf*)&sA[row0 * 32 + off0] = ldv8(&A[(size_t)(bm + row0) * K + k0 + off0]);
    *(v8bf*)&sB[row0 * 32 + off0] = ldv8(&B[(size_t)(bn + row0) * K + k0 + off0]);
    *(v8bf*)&sA[row1 * 32 + off1] = ldv8(&A[(size_t)(bm + row1) * K + k0 + off1]);
    *(v8bf*)&sB[row1 * 32 + off1] = ldv8(&B[(size_t)(bn + row1) * K + k0 + off1]);
    __syncthreads();
#endif
    const bf16_t* ar = &sA[(wave * 16 + mlo) * 32];
    v16bf afrag = cat16(ldv8(ar + kh * 16), ldv8(ar + kh * 16 + 8));
    {
      const bf16_t* br = &sB[(0 * 16 + mlo) * 32 + kh * 16];
      acc0 = wmma_bf16(afrag, cat16(ldv8(br), ldv8(br + 8)), acc0);
    }
    {
      const bf16_t* br = &sB[(1 * 16 + mlo) * 32 + kh * 16];
      acc1 = wmma_bf16(afrag, cat16(ldv8(br), ldv8(br + 8)), acc1);
    }
    {
      const bf16_t* br = &sB[(2 * 16 + mlo) * 32 + kh * 16];
      acc2 = wmma_bf16(afrag, cat16(ldv8(br), ldv8(br + 8)), acc2);
    }
    {
      const bf16_t* br = &sB[(3 * 16 + mlo) * 32 + kh * 16];
      acc3 = wmma_bf16(afrag, cat16(ldv8(br), ldv8(br + 8)), acc3);
    }
    __syncthreads();
  }
  const int mbase = bm + wave * 16 + kh * 8;
  v8f accs[4] = {acc0, acc1, acc2, acc3};
  for (int nt = 0; nt < 4; ++nt) {
    int n = bn + nt * 16 + mlo;
    for (int r = 0; r < 8; ++r) {
      size_t off = (size_t)(mbase + r) * N + n;
      float v = accs[nt][r];
      if (mode == 1) Cb[off] = (bf16_t)gelu_exact(v);
      else           Cf[off] = v;
    }
  }
}

// ---------------------------------------------------------------------------
// Per-token top-32 of 4096 scores. One block (256 thr) per token.
// ---------------------------------------------------------------------------
__global__ __launch_bounds__(256) void topk_kernel(const float* __restrict__ scores,
                                                   int* __restrict__ idx) {
  __shared__ float sc[D_FF];
  __shared__ float rv[256];
  __shared__ int   ri[256];
  const int t = blockIdx.x, tid = threadIdx.x;
  const float* row = scores + (size_t)t * D_FF;
  for (int i = tid; i < D_FF; i += 256) sc[i] = row[i];
  __syncthreads();
  for (int sel = 0; sel < K_TOP; ++sel) {
    float bv = -3.4e38f; int bi = 0;
    for (int i = tid; i < D_FF; i += 256) { float v = sc[i]; if (v > bv) { bv = v; bi = i; } }
    rv[tid] = bv; ri[tid] = bi; __syncthreads();
    for (int s = 128; s > 0; s >>= 1) {
      if (tid < s && rv[tid + s] > rv[tid]) { rv[tid] = rv[tid + s]; ri[tid] = ri[tid + s]; }
      __syncthreads();
    }
    if (tid == 0) { idx[t * K_TOP + sel] = ri[0]; sc[ri[0]] = -3.4e38f; }
    __syncthreads();
  }
}

// ---------------------------------------------------------------------------
// Fused per-token neuron attention + gate + output scatter.
// One block (128 thr, 4 waves) per token. All matmuls via WMMA bf16.
// Shared-memory pool phase-aliased.
// ---------------------------------------------------------------------------
__global__ __launch_bounds__(128) void neuron_attn_kernel(
    const float* __restrict__ x, const float* __restrict__ pats, const float* __restrict__ dirs,
    const bf16_t* __restrict__ nvec, const int* __restrict__ idx,
    const bf16_t* __restrict__ wqkv, const float* __restrict__ bqkv,
    const bf16_t* __restrict__ wout, const float* __restrict__ bout,
    const bf16_t* __restrict__ wp1,  const float* __restrict__ bp1,
    const float* __restrict__ wp2,   const float* __restrict__ bp2,
    float* __restrict__ out) {
  __shared__ __align__(16) unsigned char pool[43008];
  bf16_t* s_qk  = (bf16_t*)pool;                 // [32][256] q|k (16KB); later: s_ao [32][128]
  bf16_t* s_vt  = (bf16_t*)(pool + 16384);       // [128][32] V^T (8KB);  later: s_attended [32][128]
  bf16_t* s_sem = (bf16_t*)(pool + 24576);       // [32][128] sems; later: s_att [8][32][32]; later: s_g1 [32][64]
  int*   s_idx   = (int*)(pool + 40960);
  float* s_base  = (float*)(pool + 40960 + 128);
  float* s_final = (float*)(pool + 40960 + 256);
  float* s_red   = (float*)(pool + 40960 + 384); // 128 floats
  bf16_t* s_ao = s_qk;
  bf16_t* s_att = s_sem;
  bf16_t* s_attended = s_vt;
  bf16_t* s_g1 = s_sem;

  const int t = blockIdx.x, tid = threadIdx.x;
  const int wave = tid >> 5, lane = tid & 31, mlo = lane & 15, kh = lane >> 4;

  if (tid < K_TOP) s_idx[tid] = idx[t * K_TOP + tid];
  __syncthreads();

  // ---- gather semantics rows (bf16) + base = gelu(x . pattern) ----
  {
    const int row = tid >> 2, q = tid & 3;  // 32 rows x 4 quarters
    const bf16_t* src = nvec + (size_t)s_idx[row] * D_NEURON + q * 32;
    for (int j = 0; j < 4; ++j)
      *(v8bf*)&s_sem[row * D_NEURON + q * 32 + j * 8] = ldv8(src + j * 8);

    const float* xr = x + (size_t)t * D_MODEL + q * 256;
    const float* pr = pats + (size_t)s_idx[row] * D_MODEL + q * 256;
    float p = 0.0f;
    for (int i = 0; i < 256; i += 4) {
      float4 a = *(const float4*)(xr + i);
      float4 bb = *(const float4*)(pr + i);
      p += a.x * bb.x + a.y * bb.y + a.z * bb.z + a.w * bb.w;
    }
    s_red[tid] = p;
  }
  __syncthreads();
  if ((tid & 3) == 0) {
    int row = tid >> 2;
    float s = s_red[tid] + s_red[tid + 1] + s_red[tid + 2] + s_red[tid + 3];
    s_base[row] = gelu_exact(s);
  }
  __syncthreads();

  // ---- qkv = sems @ Wqkv^T + b : M=32, N=384, K=128 ----
  for (int nt = wave; nt < 24; nt += 4) {
    for (int mt = 0; mt < 2; ++mt) {
      v8f acc = zero_v8f();
      for (int ks = 0; ks < 4; ++ks) {
        const bf16_t* ar = &s_sem[(mt * 16 + mlo) * D_NEURON + ks * 32];
        v16bf a = cat16(ldv8(ar + kh * 16), ldv8(ar + kh * 16 + 8));
        const bf16_t* br = &wqkv[(size_t)(nt * 16 + mlo) * D_NEURON + ks * 32 + kh * 16];
        v16bf b = cat16(ldv8(br), ldv8(br + 8));
        acc = wmma_bf16(a, b, acc);
      }
      const int n = nt * 16 + mlo;
      const float bias = bqkv[n];
      for (int r = 0; r < 8; ++r) {
        int m = mt * 16 + kh * 8 + r;
        float v = acc[r] + bias;
        if (n < 256) s_qk[m * 256 + n] = (bf16_t)v;        // q | k
        else         s_vt[(n - 256) * 32 + m] = (bf16_t)v; // V transposed [dim][key]
      }
    }
  }
  __syncthreads();

  // ---- attention scores: per head S = (q @ k^T)/4, K=16 zero-padded to 32 ----
  for (int job = wave; job < 32; job += 4) {
    const int h = job >> 2, mt = (job >> 1) & 1, nt = job & 1;
    const bf16_t* ar = &s_qk[(mt * 16 + mlo) * 256 + h * HEAD_DIM + kh * 8];
    v16bf a = cat16(ldv8(ar), zero_v8bf());
    v8bf b0 = zero_v8bf(), b1 = zero_v8bf();
    {
      const bf16_t* br = &s_qk[(nt * 16 + mlo) * 256 + 128 + h * HEAD_DIM + kh * 8];
      b0 = ldv8(br);
    }
    v16bf b = cat16(b0, b1);
    v8f acc = wmma_bf16(a, b, zero_v8f());
    for (int r = 0; r < 8; ++r) {
      int qq = mt * 16 + kh * 8 + r, kk = nt * 16 + mlo;
      s_att[(h * 32 + qq) * 32 + kk] = (bf16_t)(acc[r] * 0.25f); // 1/sqrt(16)
    }
  }
  __syncthreads();

  // ---- softmax over k (rows of 32), in place, bf16 ----
  for (int r2 = tid; r2 < 8 * 32; r2 += 128) {
    bf16_t* row = &s_att[r2 * 32];
    float mx = -3.4e38f;
    for (int i = 0; i < 32; ++i) { float v = (float)row[i]; mx = v > mx ? v : mx; }
    float s = 0.0f;
    float e[32];
    for (int i = 0; i < 32; ++i) { e[i] = expf((float)row[i] - mx); s += e[i]; }
    float inv = 1.0f / s;
    for (int i = 0; i < 32; ++i) row[i] = (bf16_t)(e[i] * inv);
  }
  __syncthreads();

  // ---- ao = P @ V : per head M=32, N=16, K=32 ----
  for (int job = wave; job < 16; job += 4) {
    const int h = job >> 1, mt = job & 1;
    const bf16_t* pr = &s_att[(h * 32 + mt * 16 + mlo) * 32];
    v16bf a = cat16(ldv8(pr + kh * 16), ldv8(pr + kh * 16 + 8));
    const bf16_t* br = &s_vt[(h * HEAD_DIM + mlo) * 32 + kh * 16];
    v16bf b = cat16(ldv8(br), ldv8(br + 8));
    v8f acc = wmma_bf16(a, b, zero_v8f());
    for (int r = 0; r < 8; ++r) {
      int m = mt * 16 + kh * 8 + r;
      s_ao[m * D_NEURON + h * HEAD_DIM + mlo] = (bf16_t)acc[r];
    }
  }
  __syncthreads();

  // ---- attended = ao @ Wout^T + b : M=32, N=128, K=128 ----
  for (int job = wave; job < 16; job += 4) {
    const int nt = job >> 1, mt = job & 1;
    v8f acc = zero_v8f();
    for (int ks = 0; ks < 4; ++ks) {
      const bf16_t* ar = &s_ao[(mt * 16 + mlo) * D_NEURON + ks * 32];
      v16bf a = cat16(ldv8(ar + kh * 16), ldv8(ar + kh * 16 + 8));
      const bf16_t* br = &wout[(size_t)(nt * 16 + mlo) * D_NEURON + ks * 32 + kh * 16];
      v16bf b = cat16(ldv8(br), ldv8(br + 8));
      acc = wmma_bf16(a, b, acc);
    }
    const int n = nt * 16 + mlo;
    const float bias = bout[n];
    for (int r = 0; r < 8; ++r) {
      int m = mt * 16 + kh * 8 + r;
      s_attended[m * D_NEURON + n] = (bf16_t)(acc[r] + bias);
    }
  }
  __syncthreads();

  // ---- g1 = gelu(attended @ Wp1^T + b) : M=32, N=64, K=128 ----
  for (int job = wave; job < 8; job += 4) {
    const int nt = job >> 1, mt = job & 1;
    v8f acc = zero_v8f();
    for (int ks = 0; ks < 4; ++ks) {
      const bf16_t* ar = &s_attended[(mt * 16 + mlo) * D_NEURON + ks * 32];
      v16bf a = cat16(ldv8(ar + kh * 16), ldv8(ar + kh * 16 + 8));
      const bf16_t* br = &wp1[(size_t)(nt * 16 + mlo) * D_NEURON + ks * 32 + kh * 16];
      v16bf b = cat16(ldv8(br), ldv8(br + 8));
      acc = wmma_bf16(a, b, acc);
    }
    const int n = nt * 16 + mlo;
    const float bias = bp1[n];
    for (int r = 0; r < 8; ++r) {
      int m = mt * 16 + kh * 8 + r;
      s_g1[m * 64 + n] = (bf16_t)gelu_exact(acc[r] + bias);
    }
  }
  __syncthreads();

  // ---- gate: final = base * sigmoid(g1 . wp2 + b) ----
  if (tid < K_TOP) {
    float s = bp2[0];
    for (int i = 0; i < 64; ++i) s += (float)s_g1[tid * 64 + i] * wp2[i];
    s_final[tid] = s_base[tid] * (1.0f / (1.0f + expf(-s)));
  }
  __syncthreads();

  // ---- out[t] = sum_k final[k] * dirs[idx[k]] (dirs L2-resident) ----
  {
    const int d0 = tid * 8;
    float a0 = 0, a1 = 0, a2 = 0, a3 = 0, a4 = 0, a5 = 0, a6 = 0, a7 = 0;
    for (int k = 0; k < K_TOP; ++k) {
      const float w = s_final[k];
      const float* dr = dirs + (size_t)s_idx[k] * D_MODEL + d0;
      float4 va = *(const float4*)dr;
      float4 vb = *(const float4*)(dr + 4);
      a0 += w * va.x; a1 += w * va.y; a2 += w * va.z; a3 += w * va.w;
      a4 += w * vb.x; a5 += w * vb.y; a6 += w * vb.z; a7 += w * vb.w;
    }
    float* o = out + (size_t)t * D_MODEL + d0;
    float4 r0 = {a0, a1, a2, a3}, r1 = {a4, a5, a6, a7};
    *(float4*)o = r0; *(float4*)(o + 4) = r1;
  }
}

// ---------------------------------------------------------------------------
// Launch
// ---------------------------------------------------------------------------
extern "C" void kernel_launch(void* const* d_in, const int* in_sizes, int n_in,
                              void* d_out, int out_size, void* d_ws, size_t ws_size,
                              hipStream_t stream) {
  const float* x     = (const float*)d_in[0];
  const float* pats  = (const float*)d_in[1];
  const float* nvec  = (const float*)d_in[2];
  const float* dirs  = (const float*)d_in[3];
  const float* w1    = (const float*)d_in[4];
  const float* w2    = (const float*)d_in[5];
  const float* ln_g  = (const float*)d_in[6];
  const float* ln_b  = (const float*)d_in[7];
  const float* wqkv  = (const float*)d_in[8];
  const float* bqkv  = (const float*)d_in[9];
  const float* wout  = (const float*)d_in[10];
  const float* bout  = (const float*)d_in[11];
  const float* wp1   = (const float*)d_in[12];
  const float* bp1   = (const float*)d_in[13];
  const float* wp2   = (const float*)d_in[14];
  const float* bp2   = (const float*)d_in[15];
  float* out = (float*)d_out;

  const int NTOK = in_sizes[0] / D_MODEL;  // 4096

  // Workspace carve (bytes)
  unsigned char* ws = (unsigned char*)d_ws;
  size_t off = 0;
  bf16_t* xn_bf   = (bf16_t*)(ws + off); off += (size_t)NTOK * D_MODEL * 2;
  bf16_t* h_bf    = (bf16_t*)(ws + off); off += (size_t)NTOK * D_MODEL * 2;
  float*  scores  = (float*) (ws + off); off += (size_t)NTOK * D_FF * 4;
  int*    idxbuf  = (int*)   (ws + off); off += (size_t)NTOK * K_TOP * 4;
  bf16_t* w1_bf   = (bf16_t*)(ws + off); off += (size_t)D_MODEL * D_MODEL * 2;
  bf16_t* w2_bf   = (bf16_t*)(ws + off); off += (size_t)D_FF * D_MODEL * 2;
  bf16_t* wqkv_bf = (bf16_t*)(ws + off); off += (size_t)3 * D_NEURON * D_NEURON * 2;
  bf16_t* wout_bf = (bf16_t*)(ws + off); off += (size_t)D_NEURON * D_NEURON * 2;
  bf16_t* wp1_bf  = (bf16_t*)(ws + off); off += (size_t)(D_NEURON / 2) * D_NEURON * 2;
  bf16_t* nvec_bf = (bf16_t*)(ws + off); off += (size_t)D_FF * D_NEURON * 2;

  cast_f32_bf16<<<(D_MODEL * D_MODEL + 255) / 256, 256, 0, stream>>>(w1, w1_bf, D_MODEL * D_MODEL);
  cast_f32_bf16<<<(D_FF * D_MODEL + 255) / 256, 256, 0, stream>>>(w2, w2_bf, D_FF * D_MODEL);
  cast_f32_bf16<<<(3 * D_NEURON * D_NEURON + 255) / 256, 256, 0, stream>>>(wqkv, wqkv_bf, 3 * D_NEURON * D_NEURON);
  cast_f32_bf16<<<(D_NEURON * D_NEURON + 255) / 256, 256, 0, stream>>>(wout, wout_bf, D_NEURON * D_NEURON);
  cast_f32_bf16<<<((D_NEURON / 2) * D_NEURON + 255) / 256, 256, 0, stream>>>(wp1, wp1_bf, (D_NEURON / 2) * D_NEURON);
  cast_f32_bf16<<<(D_FF * D_NEURON + 255) / 256, 256, 0, stream>>>(nvec, nvec_bf, D_FF * D_NEURON);

  ln_kernel<<<NTOK, 256, 0, stream>>>(x, ln_g, ln_b, xn_bf);

  // h = gelu(xn @ W1^T)  [NTOK x 1024]
  gemm_bf16_kernel<<<dim3(D_MODEL / 64, NTOK / 64), 128, 0, stream>>>(
      xn_bf, w1_bf, (float*)0, h_bf, NTOK, D_MODEL, D_MODEL, 1);

  // scores = h @ W2^T    [NTOK x 4096]
  gemm_bf16_kernel<<<dim3(D_FF / 64, NTOK / 64), 128, 0, stream>>>(
      h_bf, w2_bf, scores, (bf16_t*)0, NTOK, D_FF, D_MODEL, 0);

  topk_kernel<<<NTOK, 256, 0, stream>>>(scores, idxbuf);

  neuron_attn_kernel<<<NTOK, 128, 0, stream>>>(
      x, pats, dirs, nvec_bf, idxbuf,
      wqkv_bf, bqkv, wout_bf, bout, wp1_bf, bp1, wp2, bp2, out);

  (void)n_in; (void)out_size; (void)ws_size;
}
